// SemanticGraphBlock_78288663872183
// MI455X (gfx1250) — matrix operand order repaired and checked by reading, not compiled, not run on hardware
//
#include <hip/hip_runtime.h>

typedef __attribute__((ext_vector_type(16))) __bf16 v16bf;
typedef __attribute__((ext_vector_type(8)))  float  v8f;

constexpr int NN = 8192;   // nodes
constexpr int DD = 512;    // in dim
constexpr int HH = 256;    // hidden
constexpr int EE = 262144; // edges
constexpr float F_ALPHA = 0.6f;

// ---------------- workspace layout (bytes) ----------------
constexpr size_t OFF_DEG   = 0;
constexpr size_t SZ_DEG    = (size_t)NN * 4;                 // 32 KB
constexpr size_t OFF_AGG   = OFF_DEG + SZ_DEG;
constexpr size_t SZ_AGG    = (size_t)NN * DD * 4;            // 16 MB
constexpr size_t OFF_BMP   = OFF_AGG + SZ_AGG;
constexpr size_t SZ_BMP    = (size_t)NN * (NN / 32) * 4;     // 8 MB
constexpr size_t OFF_A1    = OFF_BMP + SZ_BMP;
constexpr size_t SZ_A1     = (size_t)NN * (2 * DD) * 2;      // 16 MB bf16
constexpr size_t OFF_W1T   = OFF_A1 + SZ_A1;
constexpr size_t SZ_W1T    = (size_t)(2 * DD) * HH * 2;      // 512 KB
constexpr size_t OFF_W2T   = OFF_W1T + SZ_W1T;
constexpr size_t SZ_W2T    = (size_t)(2 * HH) * HH * 2;      // 256 KB
constexpr size_t OFF_H     = OFF_W2T + SZ_W2T;
constexpr size_t SZ_H      = (size_t)NN * HH * 4;            // 8 MB
constexpr size_t OFF_A2    = OFF_H + SZ_H;
constexpr size_t SZ_A2     = (size_t)NN * (2 * HH) * 2;      // 8 MB bf16
constexpr size_t OFF_PROBS = OFF_A2 + SZ_A2;
constexpr size_t SZ_PROBS  = (size_t)NN * 5 * 4;
constexpr size_t OFF_P     = OFF_PROBS + SZ_PROBS;
constexpr size_t SZ_P      = (size_t)5 * HH * 4;
constexpr size_t OFF_S     = OFF_P + SZ_P;                   // 5 floats (pad 64)
constexpr size_t OFF_MAX   = OFF_S + 64;                     // 1 u32

// ---------------- utility ----------------
__global__ void zero_u32(unsigned* __restrict__ p, size_t n) {
  for (size_t i = (size_t)blockIdx.x * blockDim.x + threadIdx.x; i < n;
       i += (size_t)gridDim.x * blockDim.x)
    p[i] = 0u;
}

// deg (with duplicates, segment_sum semantics) + dedup adjacency bitmap (.set semantics)
__global__ void edge_scatter(const int* __restrict__ row, const int* __restrict__ col,
                             float* __restrict__ deg, unsigned* __restrict__ bmp) {
  int e = blockIdx.x * blockDim.x + threadIdx.x;
  if (e >= EE) return;
  int r = row[e], c = col[e];
  atomicAdd(&deg[r], 1.0f);
  atomicOr(&bmp[(size_t)r * (NN / 32) + (c >> 5)], 1u << (c & 31));
}

// segment_sum(x[col], row): each thread scatters 4 floats of one edge
__global__ void agg_scatter(const float* __restrict__ x, const int* __restrict__ row,
                            const int* __restrict__ col, float* __restrict__ aggsum) {
  size_t tid = (size_t)blockIdx.x * blockDim.x + threadIdx.x;
  if (tid >= (size_t)EE * (DD / 4)) return;
  size_t e = tid >> 7;            // DD/4 == 128
  int q = (int)(tid & 127);
  int r = row[e], c = col[e];
  float4 v = reinterpret_cast<const float4*>(x)[(size_t)c * (DD / 4) + q];
  float* dst = aggsum + (size_t)r * DD + q * 4;
  atomicAdd(dst + 0, v.x); atomicAdd(dst + 1, v.y);
  atomicAdd(dst + 2, v.z); atomicAdd(dst + 3, v.w);
}

// A1 = bf16(concat(x, aggsum/deg))   (8192 x 1024)
__global__ void build_a1(const float* __restrict__ x, const float* __restrict__ deg,
                         const float* __restrict__ aggsum, __bf16* __restrict__ A1) {
  size_t tid = (size_t)blockIdx.x * blockDim.x + threadIdx.x;
  if (tid >= (size_t)NN * DD) return;
  size_t i = tid >> 9;            // DD == 512
  int d = (int)(tid & (DD - 1));
  float dg = fmaxf(deg[i], 1.0f);
  A1[i * (2 * DD) + d]      = (__bf16)x[i * DD + d];
  A1[i * (2 * DD) + DD + d] = (__bf16)(aggsum[i * DD + d] / dg);
}

// Wt[n*K + k] = bf16(W[k*Nc + n])
__global__ void transpose_w(const float* __restrict__ W, __bf16* __restrict__ Wt,
                            int K, int Nc) {
  int tid = blockIdx.x * blockDim.x + threadIdx.x;
  if (tid >= K * Nc) return;
  int k = tid / Nc, n = tid - k * Nc;
  Wt[(size_t)n * K + k] = (__bf16)W[tid];
}

// ---------------- WMMA GEMM: (M x K bf16) @ (K x 256 bf16) + bias, relu(l2norm) ----
// block = 512 threads = 16 waves; block owns one 16-row M-tile x all 256 cols.
// The shared 16xK A-tile is staged in LDS once via async global->LDS DMA loads
// (ASYNCcnt path), killing the 16x redundant global reads across waves.
__global__ __launch_bounds__(512) void gemm_rowblock(
    const __bf16* __restrict__ A, const __bf16* __restrict__ Bt,
    const float* __restrict__ bias, int K,
    float* __restrict__ outF, __bf16* __restrict__ outB, int obStride) {
  __shared__ __align__(16) char asmem[16 * 1024 * 2];   // up to K=1024: 32 KB
  __shared__ float rowsq[16];

  const int wave = threadIdx.x >> 5;
  const int lane = threadIdx.x & 31;
  const int half = lane >> 4;
  const int ln   = lane & 15;
  const int m0   = blockIdx.x << 4;
  const int n0   = wave << 4;

  if (threadIdx.x < 16) rowsq[threadIdx.x] = 0.0f;

  // ---- async stage of the contiguous 16-row A block (16*K bf16) into LDS ----
  {
    const char* gsrc = (const char*)(A + (size_t)m0 * K);
    const size_t blockBytes = (size_t)16 * K * 2;       // 32 KB (K=1024) / 16 KB (K=512)
    for (size_t off = (size_t)threadIdx.x * 16; off < blockBytes; off += 512 * 16) {
      unsigned ldsaddr = (unsigned)(uintptr_t)(asmem + off);   // addr[31:0] == LDS offset
      unsigned long long ga = (unsigned long long)(uintptr_t)(gsrc + off);
      asm volatile("global_load_async_to_lds_b128 %0, %1, off"
                   :: "v"(ldsaddr), "v"(ga) : "memory");
    }
    asm volatile("s_wait_asynccnt 0" ::: "memory");
    __syncthreads();
  }

  const __bf16* brow = Bt + (size_t)(n0 + ln) * K;
  const char* arow = asmem + (size_t)ln * K * 2;

  v8f acc = {};
  for (int kb = 0; kb < K; kb += 32) {
    union { uint4 q[2]; v16bf v; } fa, fb;
    // A 16x32 bf16 fragment from LDS: lane-half -> K window {0..7,16..23}/{8..15,24..31}
    const uint4* ap = reinterpret_cast<const uint4*>(arow + (size_t)(kb + half * 8) * 2);
    fa.q[0] = ap[0];
    fa.q[1] = ap[2];                       // +16 elements
    // B 32x16 bf16: lane-half -> K window {0..7,16..23}/{8..15,24..31} for column ln
    const uint4* bp = reinterpret_cast<const uint4*>(brow + kb + half * 8);
    fb.q[0] = bp[0];
    fb.q[1] = bp[2];
    __builtin_prefetch(brow + kb + 64, 0, 3);
    acc = __builtin_amdgcn_wmma_f32_16x16x32_bf16(
        false, fa.v, false, fb.v, (short)0, acc, false, false);
  }

  __syncthreads();

  const float b = bias[n0 + ln];
  float vals[8];
#pragma unroll
  for (int e = 0; e < 8; ++e) {           // C/D: VGPR e -> row half*8+e, col ln
    float v = acc[e] + b;
    vals[e] = v;
    atomicAdd(&rowsq[half * 8 + e], v * v);
  }
  __syncthreads();

#pragma unroll
  for (int e = 0; e < 8; ++e) {
    int r = half * 8 + e;
    float scale = rsqrtf(fmaxf(rowsq[r], 1e-12f));
    float o = fmaxf(vals[e] * scale, 0.0f);   // relu(l2norm)
    outF[(size_t)(m0 + r) * HH + n0 + ln] = o;
    if (outB) outB[(size_t)(m0 + r) * obStride + n0 + ln] = (__bf16)o;
  }
}

// logits = h@Wl+bl, softmax, accumulate s-vector + max ||p||^2 (== max class_sim,
// attained on the diagonal by Cauchy-Schwarz)
__global__ void logits_softmax(const float* __restrict__ h, const float* __restrict__ Wl,
                               const float* __restrict__ bl, float* __restrict__ probs,
                               float* __restrict__ svec, unsigned* __restrict__ maxbits) {
  int node = (blockIdx.x * blockDim.x + threadIdx.x) >> 5;
  int lane = threadIdx.x & 31;
  if (node >= NN) return;
  float acc[5] = {0.f, 0.f, 0.f, 0.f, 0.f};
  for (int c = lane; c < HH; c += 32) {
    float hv = h[(size_t)node * HH + c];
#pragma unroll
    for (int cl = 0; cl < 5; ++cl) acc[cl] += hv * Wl[c * 5 + cl];
  }
#pragma unroll
  for (int cl = 0; cl < 5; ++cl)
    for (int off = 16; off; off >>= 1) acc[cl] += __shfl_xor(acc[cl], off, 32);
  if (lane == 0) {
    float m = -3.4e38f;
#pragma unroll
    for (int cl = 0; cl < 5; ++cl) { acc[cl] += bl[cl]; m = fmaxf(m, acc[cl]); }
    float ssum = 0.f, p[5];
#pragma unroll
    for (int cl = 0; cl < 5; ++cl) { p[cl] = __expf(acc[cl] - m); ssum += p[cl]; }
    float n2 = 0.f;
#pragma unroll
    for (int cl = 0; cl < 5; ++cl) {
      p[cl] /= ssum;
      probs[(size_t)node * 5 + cl] = p[cl];
      atomicAdd(&svec[cl], p[cl]);
      n2 += p[cl] * p[cl];
    }
    atomicMax(maxbits, __float_as_uint(n2));   // values > 0: bit order == value order
  }
}

// P = probs^T @ h  (5 x 256)
__global__ void pmat(const float* __restrict__ h, const float* __restrict__ probs,
                     float* __restrict__ P) {
  int t = threadIdx.x;
  float acc[5] = {0.f, 0.f, 0.f, 0.f, 0.f};
  for (int i = blockIdx.x; i < NN; i += gridDim.x) {
    float hv = h[(size_t)i * HH + t];
#pragma unroll
    for (int cl = 0; cl < 5; ++cl) acc[cl] += probs[(size_t)i * 5 + cl] * hv;
  }
#pragma unroll
  for (int cl = 0; cl < 5; ++cl) atomicAdd(&P[cl * HH + t], acc[cl]);
}

// agg2[i] = (a*adj@h + c*probs[i]@P) / (a*deg_u + c*probs[i].s);  A2[:,256:512] = bf16(agg2)
__global__ void adj_agg(const float* __restrict__ h, const unsigned* __restrict__ bmp,
                        const float* __restrict__ probs, const float* __restrict__ P,
                        const float* __restrict__ svec, const unsigned* __restrict__ maxbits,
                        __bf16* __restrict__ A2) {
  const int i = blockIdx.x, t = threadIdx.x;
  __shared__ unsigned list[1024];
  __shared__ unsigned cnt;
  if (t == 0) cnt = 0;
  __syncthreads();
  unsigned w = bmp[(size_t)i * (NN / 32) + t];
  unsigned n = __popc(w);
  unsigned base = n ? atomicAdd(&cnt, n) : 0u;
  while (w) {
    unsigned b = (unsigned)__builtin_ctz(w);
    w &= w - 1;
    if (base < 1024) list[base] = (t << 5) + b;
    ++base;
  }
  __syncthreads();
  unsigned total = cnt;
  unsigned m = total < 1024u ? total : 1024u;
  float acc = 0.f;
  for (unsigned l = 0; l < m; ++l) acc += h[(size_t)list[l] * HH + t];

  float pc[5], pdS = 0.f, pdP = 0.f;
#pragma unroll
  for (int cl = 0; cl < 5; ++cl) {
    pc[cl] = probs[(size_t)i * 5 + cl];
    pdS += pc[cl] * svec[cl];
    pdP += pc[cl] * P[cl * HH + t];
  }
  float csc = (1.0f - F_ALPHA) / __uint_as_float(*maxbits);
  float num = F_ALPHA * acc + csc * pdP;
  float den = F_ALPHA * (float)total + csc * pdS;
  A2[(size_t)i * (2 * HH) + HH + t] = (__bf16)(num / fmaxf(den, 1e-12f));
}

extern "C" void kernel_launch(void* const* d_in, const int* in_sizes, int n_in,
                              void* d_out, int out_size, void* d_ws, size_t ws_size,
                              hipStream_t stream) {
  const float* x   = (const float*)d_in[0];
  const int*   ei  = (const int*)d_in[1];
  const float* W1  = (const float*)d_in[2];
  const float* b1  = (const float*)d_in[3];
  const float* Wl  = (const float*)d_in[4];
  const float* bl  = (const float*)d_in[5];
  const float* W2  = (const float*)d_in[6];
  const float* b2  = (const float*)d_in[7];
  const int* row = ei;
  const int* col = ei + EE;

  char* ws = (char*)d_ws;
  float*    deg    = (float*)(ws + OFF_DEG);
  float*    aggsum = (float*)(ws + OFF_AGG);
  unsigned* bmp    = (unsigned*)(ws + OFF_BMP);
  __bf16*   A1     = (__bf16*)(ws + OFF_A1);
  __bf16*   W1t    = (__bf16*)(ws + OFF_W1T);
  __bf16*   W2t    = (__bf16*)(ws + OFF_W2T);
  float*    h      = (float*)(ws + OFF_H);
  __bf16*   A2     = (__bf16*)(ws + OFF_A2);
  float*    probs  = (float*)(ws + OFF_PROBS);
  float*    P      = (float*)(ws + OFF_P);
  float*    svec   = (float*)(ws + OFF_S);
  unsigned* maxb   = (unsigned*)(ws + OFF_MAX);

  // zero: deg + aggsum + bitmap (contiguous) ; P + s + max
  zero_u32<<<4096, 256, 0, stream>>>((unsigned*)ws, OFF_A1 / 4);
  zero_u32<<<8, 256, 0, stream>>>((unsigned*)(ws + OFF_P), (SZ_P + 128) / 4);

  edge_scatter<<<(EE + 255) / 256, 256, 0, stream>>>(row, col, deg, bmp);
  agg_scatter<<<(int)(((size_t)EE * (DD / 4) + 255) / 256), 256, 0, stream>>>(x, row, col, aggsum);
  build_a1<<<(int)(((size_t)NN * DD + 255) / 256), 256, 0, stream>>>(x, deg, aggsum, A1);
  transpose_w<<<(2 * DD * HH + 255) / 256, 256, 0, stream>>>(W1, W1t, 2 * DD, HH);
  transpose_w<<<(2 * HH * HH + 255) / 256, 256, 0, stream>>>(W2, W2t, 2 * HH, HH);

  // layer 1: h = relu(l2norm(A1 @ W1 + b1));   also writes bf16 h into A2[:, 0:256]
  gemm_rowblock<<<NN / 16, 512, 0, stream>>>(A1, W1t, b1, 2 * DD, h, A2, 2 * HH);

  logits_softmax<<<NN / 8, 256, 0, stream>>>(h, Wl, bl, probs, svec, maxb);
  pmat<<<64, 256, 0, stream>>>(h, probs, P);
  adj_agg<<<NN, 256, 0, stream>>>(h, bmp, probs, P, svec, maxb, A2);

  // layer 2: out = relu(l2norm(A2 @ W2 + b2))
  gemm_rowblock<<<NN / 16, 512, 0, stream>>>(A2, W2t, b2, 2 * HH, (float*)d_out, nullptr, 0);
}
